// GCN_12953621364999
// MI455X (gfx1250) — matrix operand order repaired and checked by reading, not compiled, not run on hardware
//
#include <hip/hip_runtime.h>
#include <hip/hip_bf16.h>

// ---------------------------------------------------------------------------
// Types
// ---------------------------------------------------------------------------
typedef __attribute__((ext_vector_type(4)))  float   f4;
typedef __attribute__((ext_vector_type(2)))  float   f2;
typedef __attribute__((ext_vector_type(8)))  float   v8f;
typedef __attribute__((ext_vector_type(16))) __bf16  v16bf;
typedef __attribute__((ext_vector_type(8)))  __bf16  v8bf;
typedef __attribute__((ext_vector_type(4)))  __bf16  v4bf;
typedef __attribute__((ext_vector_type(4)))  int     v4i;

// Problem constants (fixed by the reference)
#define D_DIA  128
#define L_DIA  40
#define N_DIM  512
#define NHID   512
#define N_NODE 15360            // 3 * 128 * 40
#define INV_DEG (1.0f / 42.0f)  // uniform degree: 39 intra + 2 cross + 1 self
#define WSZ    (NHID * NHID)    // one 512x512 weight matrix

// GEMM tiling
#define BM 128
#define BN 128
#define BK 64
#define LDT 80   // padded bf16 k-stride of LDS tiles (160B rows, 16B aligned)

// ---------------------------------------------------------------------------
// Async global->LDS 16-byte copy (gfx1250 GLOBAL_LOAD_ASYNC_TO_LDS_B128).
// ---------------------------------------------------------------------------
#if __has_builtin(__builtin_amdgcn_global_load_async_to_lds_b128)
#define HAS_ASYNC_LDS 1
#endif

__device__ __forceinline__ void copy16_g2l(const __bf16* g, __bf16* l) {
#if defined(HAS_ASYNC_LDS)
  __builtin_amdgcn_global_load_async_to_lds_b128(
      (__attribute__((address_space(1))) v4i*)(v4i*)g,
      (__attribute__((address_space(3))) v4i*)(v4i*)l,
      0, 0);
#else
  *(v8bf*)l = *(const v8bf*)g;
#endif
}

__device__ __forceinline__ void wait_async_copies() {
#if defined(HAS_ASYNC_LDS)
#if __has_builtin(__builtin_amdgcn_s_wait_asynccnt)
  __builtin_amdgcn_s_wait_asynccnt(0);
#else
  asm volatile("s_wait_asynccnt 0" ::: "memory");
#endif
#endif
}

// ---------------------------------------------------------------------------
// One-time per launch: Wt[g][n][k] = bf16(W[g][k][n]); g=0 -> fc1_W,
// g=1..4 -> convW[g-1]. Pre-transposed so GEMM B staging is a block copy.
// ---------------------------------------------------------------------------
__global__ __launch_bounds__(256) void transpose_weights(
    const float* __restrict__ fc1W, const float* __restrict__ convW,
    __bf16* __restrict__ Wt)
{
  __shared__ __bf16 tile[64][72];        // 144B rows, 16B aligned
  const int g  = blockIdx.z;
  const int kt = blockIdx.x * 64;
  const int nt = blockIdx.y * 64;
  const float* W = (g == 0) ? fc1W : (convW + (size_t)(g - 1) * WSZ);

  const int tid = threadIdx.x;
  const int row = tid >> 2;              // k within tile: 0..63
  const int cb  = (tid & 3) * 16;        // n within tile
  #pragma unroll
  for (int vv = 0; vv < 4; ++vv) {
    f4 d = *(const f4*)(W + (size_t)(kt + row) * NHID + nt + cb + vv * 4);
    tile[cb + vv * 4 + 0][row] = (__bf16)d.x;
    tile[cb + vv * 4 + 1][row] = (__bf16)d.y;
    tile[cb + vv * 4 + 2][row] = (__bf16)d.z;
    tile[cb + vv * 4 + 3][row] = (__bf16)d.w;
  }
  __syncthreads();

  const int nl   = tid >> 2;             // n within tile: 0..63
  const int kseg = (tid & 3) * 16;       // k within tile
  __bf16* dst = Wt + (size_t)g * WSZ + (size_t)(nt + nl) * NHID + kt + kseg;
  *(v8bf*)dst       = *(v8bf*)&tile[nl][kseg];
  *(v8bf*)(dst + 8) = *(v8bf*)&tile[nl][kseg + 8];
}

// ---------------------------------------------------------------------------
// feats[node] = {l+spk, a, v} per dialogue as [l(40), a(40), v(40)];
// writes fp32 (kept for output concat) and a bf16 mirror (GEMM A input).
// ---------------------------------------------------------------------------
__global__ __launch_bounds__(128) void build_feats(
    const float* __restrict__ a, const float* __restrict__ v,
    const float* __restrict__ l, const float* __restrict__ qmask,
    const float* __restrict__ spk, float* __restrict__ feats,
    __bf16* __restrict__ featsB)
{
  const int node = blockIdx.x;        // 0..15359
  const int c4   = threadIdx.x;       // 0..127 (float4 lanes of 512 features)
  const int d = node / 120;
  const int r = node % 120;
  const int m = r / 40;
  const int t = r % 40;
  const int u = d * 40 + t;

  const float* src = (m == 0) ? l : ((m == 1) ? a : v);
  f4 val = *(const f4*)(src + (size_t)u * N_DIM + c4 * 4);
  if (m == 0) {
    const int qi = (t * D_DIA + d) * 2;          // qmask[t, d, :]
    const int s  = (qmask[qi + 1] > qmask[qi]) ? 1 : 0;
    f4 e = *(const f4*)(spk + s * N_DIM + c4 * 4);
    val.x += e.x; val.y += e.y; val.z += e.z; val.w += e.w;
  }
  *(f4*)(feats + (size_t)node * N_DIM + c4 * 4) = val;
  v4bf h;
  h[0] = (__bf16)val.x; h[1] = (__bf16)val.y;
  h[2] = (__bf16)val.z; h[3] = (__bf16)val.w;
  *(v4bf*)(featsB + (size_t)node * N_DIM + c4 * 4) = h;
}

// ---------------------------------------------------------------------------
// bsum[d*3+m] = sum over 40 rows of gnn block (closed-form intra-modal sum)
// ---------------------------------------------------------------------------
__global__ __launch_bounds__(256) void block_sums(
    const float* __restrict__ gnn, float* __restrict__ bsum)
{
  const int blk = blockIdx.x;          // 0..383
  const int c   = threadIdx.x * 2;     // column pair
  const float* base = gnn + (size_t)blk * 40 * NHID + c;
  float s0 = 0.f, s1 = 0.f;
  #pragma unroll 8
  for (int t = 0; t < 40; ++t) {
    f2 x = *(const f2*)(base + (size_t)t * NHID);
    s0 += x.x; s1 += x.y;
  }
  bsum[blk * NHID + c]     = s0;
  bsum[blk * NHID + c + 1] = s1;
}

// ---------------------------------------------------------------------------
// aggB(d,m,t) = bf16((blocksum(d,m) + gnn(d,m+1,t) + gnn(d,m+2,t)) / 42)
// bf16 at rest: this buffer is only ever a GEMM A operand.
// ---------------------------------------------------------------------------
__global__ __launch_bounds__(256) void build_agg(
    const float* __restrict__ gnn, const float* __restrict__ bsum,
    __bf16* __restrict__ aggB)
{
  const size_t gid = (size_t)blockIdx.x * 256 + threadIdx.x; // float4 id
  const int node = (int)(gid >> 7);
  const int c4   = (int)(gid & 127);
  const int d = node / 120;
  const int r = node % 120;
  const int m = r / 40;
  const int t = r % 40;
  const int n1 = d * 120 + ((m + 1) % 3) * 40 + t;
  const int n2 = d * 120 + ((m + 2) % 3) * 40 + t;

  f4 s  = *(const f4*)(bsum + (size_t)(d * 3 + m) * NHID + c4 * 4);
  f4 xa = *(const f4*)(gnn + (size_t)n1 * NHID + c4 * 4);
  f4 xb = *(const f4*)(gnn + (size_t)n2 * NHID + c4 * 4);
  v4bf o;
  o[0] = (__bf16)((s.x + xa.x + xb.x) * INV_DEG);
  o[1] = (__bf16)((s.y + xa.y + xb.y) * INV_DEG);
  o[2] = (__bf16)((s.z + xa.z + xb.z) * INV_DEG);
  o[3] = (__bf16)((s.w + xa.w + xb.w) * INV_DEG);
  *(v4bf*)(aggB + gid * 4) = o;
}

// ---------------------------------------------------------------------------
// C[M,N] (+)= A[M,K] @ Wt[N,K]^T + bias ; optional mirror store to C2.
// A, Wt are bf16 in HBM; double-buffered async b128 staging into LDS
// (ASYNCcnt-pipelined: tile i+1 streams in while tile i is in the WMMAs);
// compute with v_wmma_f32_16x16x32_bf16; fp32 accumulate/epilogue.
// Block tile 128x128, BK=64, 8 waves (2x4), wave tile 64x32.
// ---------------------------------------------------------------------------
__global__ __launch_bounds__(256) void gemm_bf16_wmma(
    const __bf16* __restrict__ A, const __bf16* __restrict__ Wt,
    const float* __restrict__ bias, float* __restrict__ C,
    float* __restrict__ C2, int M, int N, int K, int accumulate)
{
  __shared__ __bf16 As[2][BM * LDT];   // 2 x 20KB
  __shared__ __bf16 Bs[2][BN * LDT];   // 2 x 20KB  (80KB total, 320KB WGP LDS)

  const int tid  = threadIdx.x;
  const int lane = tid & 31;
  const int wave = tid >> 5;
  const int bm = blockIdx.x * BM;
  const int bn = blockIdx.y * BN;

  const int wm = (wave >> 2) * 64;   // 2 waves along M
  const int wn = (wave & 3) * 32;    // 4 waves along N

  const int lrow = lane & 15;        // row/col within 16-wide fragment
  const int lg   = lane >> 4;        // half-wave group

  v8f acc[4][2];
  #pragma unroll
  for (int i = 0; i < 4; ++i)
    #pragma unroll
    for (int j = 0; j < 2; ++j)
      acc[i][j] = (v8f)0.f;

  // staging decomposition: 1024 16B chunks per tile, 4 per thread
  const int srow = tid >> 1;                    // shared by the 4 chunks? no:
  (void)srow;

  auto stage = [&](int buf, int k0) {
    #pragma unroll
    for (int i = 0; i < 4; ++i) {
      const int q   = tid + i * 256;
      const int row = q >> 3;            // 0..127
      const int kb  = (q & 7) * 8;       // bf16 offset within row: 0..56
      copy16_g2l(A  + (size_t)(bm + row) * K + k0 + kb,
                 &As[buf][row * LDT + kb]);
      copy16_g2l(Wt + (size_t)(bn + row) * K + k0 + kb,
                 &Bs[buf][row * LDT + kb]);
    }
  };

  // prologue: fill buffer 0
  stage(0, 0);
  wait_async_copies();
  __syncthreads();

  const int ntiles = K / BK;
  for (int it = 0; it < ntiles; ++it) {
    const int cur = it & 1;
    const int nxt = cur ^ 1;

    // kick off next tile's async copies; they overlap the WMMAs below
    if (it + 1 < ntiles) stage(nxt, (it + 1) * BK);

    // ---- compute from buffer `cur`: 2 k-steps of 32, 8 WMMAs each ----
    #pragma unroll
    for (int kk = 0; kk < BK; kk += 32) {
      v16bf bfrag[2];
      #pragma unroll
      for (int ni = 0; ni < 2; ++ni) {
        // B 32x16 bf16 layout: lane holds col (lane&15),
        // k chunks [8g,8g+8) and [16+8g,16+8g+8)
        const __bf16* bp = &Bs[cur][(wn + ni * 16 + lrow) * LDT + kk];
        v8bf lo = *(const v8bf*)(bp + lg * 8);
        v8bf hi = *(const v8bf*)(bp + 16 + lg * 8);
        bfrag[ni] = __builtin_shufflevector(lo, hi,
            0,1,2,3,4,5,6,7,8,9,10,11,12,13,14,15);
      }
      #pragma unroll
      for (int mi = 0; mi < 4; ++mi) {
        // A 16x32 bf16 layout: lane holds row (lane&15),
        // k chunks [8g,8g+8) and [16+8g,16+8g+8)
        const __bf16* ap = &As[cur][(wm + mi * 16 + lrow) * LDT + kk];
        v8bf lo = *(const v8bf*)(ap + lg * 8);
        v8bf hi = *(const v8bf*)(ap + 16 + lg * 8);
        v16bf afrag = __builtin_shufflevector(lo, hi,
            0,1,2,3,4,5,6,7,8,9,10,11,12,13,14,15);
        acc[mi][0] = __builtin_amdgcn_wmma_f32_16x16x32_bf16(
            false, afrag, false, bfrag[0], (short)0, acc[mi][0], false, false);
        acc[mi][1] = __builtin_amdgcn_wmma_f32_16x16x32_bf16(
            false, afrag, false, bfrag[1], (short)0, acc[mi][1], false, false);
      }
    }

    // drain this wave's async copies, then make them visible block-wide
    wait_async_copies();
    __syncthreads();
  }

  // ---- epilogue: bias, optional residual accumulate, optional mirror ----
  #pragma unroll
  for (int mi = 0; mi < 4; ++mi) {
    #pragma unroll
    for (int ni = 0; ni < 2; ++ni) {
      const int col  = bn + wn + ni * 16 + lrow;
      const int row0 = bm + wm + mi * 16 + lg * 8;
      const float bv = bias[col];
      #pragma unroll
      for (int j = 0; j < 8; ++j) {
        const size_t idx = (size_t)(row0 + j) * N + col;
        float rres = acc[mi][ni][j] + bv;
        if (accumulate) rres += C[idx];
        C[idx] = rres;
        if (C2) C2[idx] = rres;
      }
    }
  }
}

// ---------------------------------------------------------------------------
// out1[u, m*1536 + j] = concat(feats, x1, gnn)[node(d,m,t), j]  (reverse feat.)
// ---------------------------------------------------------------------------
__global__ __launch_bounds__(256) void write_out(
    const float* __restrict__ feats, const float* __restrict__ x1,
    const float* __restrict__ gnn, float* __restrict__ out)
{
  const size_t gid = (size_t)blockIdx.x * 256 + threadIdx.x; // float4 id
  const int u   = (int)(gid / 1152);       // 4608/4
  const int rem = (int)(gid % 1152);
  const int m   = rem / 384;
  const int j4  = rem % 384;
  const int d = u / 40, t = u % 40;
  const int n = d * 120 + m * 40 + t;
  const float* src = (j4 < 128) ? feats : ((j4 < 256) ? x1 : gnn);
  const int jj = j4 & 127;
  f4 val = *(const f4*)(src + (size_t)n * NHID + jj * 4);
  *(f4*)(out + gid * 4) = val;
}

// ---------------------------------------------------------------------------
extern "C" void kernel_launch(void* const* d_in, const int* in_sizes, int n_in,
                              void* d_out, int out_size, void* d_ws, size_t ws_size,
                              hipStream_t stream) {
  const float* a      = (const float*)d_in[0];
  const float* v      = (const float*)d_in[1];
  const float* l      = (const float*)d_in[2];
  const float* qmask  = (const float*)d_in[3];
  const float* spk    = (const float*)d_in[4];
  const float* fc1_W  = (const float*)d_in[5];
  const float* fc1_b  = (const float*)d_in[6];
  const float* convW  = (const float*)d_in[7];
  const float* convb  = (const float*)d_in[8];
  float* out = (float*)d_out;

  const size_t MAT = (size_t)N_NODE * NHID;   // 7,864,320 elements
  char* ws = (char*)d_ws;
  float*  feats  = (float*)ws;                 ws += MAT * 4;
  float*  x1     = (float*)ws;                 ws += MAT * 4;
  float*  gnn    = (float*)ws;                 ws += MAT * 4;
  __bf16* featsB = (__bf16*)ws;                ws += MAT * 2;
  __bf16* aggB   = (__bf16*)ws;                ws += MAT * 2;
  __bf16* WtB    = (__bf16*)ws;                ws += (size_t)5 * WSZ * 2;
  float*  bsum   = (float*)ws;                 // 384*512 floats

  // 0) convert + transpose all 5 weight matrices to bf16 [n][k]
  transpose_weights<<<dim3(8, 8, 5), 256, 0, stream>>>(fc1_W, convW, WtB);

  // 1) node features (+ speaker embedding on text modality), fp32 + bf16
  build_feats<<<N_NODE, 128, 0, stream>>>(a, v, l, qmask, spk, feats, featsB);

  // 2) x1 = feats @ fc1_W + b ; gnn initialized to x1 via mirror store
  dim3 ggrid(N_NODE / BM, NHID / BN);
  gemm_bf16_wmma<<<ggrid, 256, 0, stream>>>(featsB, WtB, fc1_b, x1, gnn,
                                            N_NODE, NHID, N_DIM, 0);

  // 3) four GCN layers: gnn += gcn_conv(gnn)  (closed-form aggregation)
  for (int k = 0; k < 4; ++k) {
    block_sums<<<384, 256, 0, stream>>>(gnn, bsum);
    build_agg<<<(N_NODE * (NHID / 4)) / 256, 256, 0, stream>>>(gnn, bsum, aggB);
    gemm_bf16_wmma<<<ggrid, 256, 0, stream>>>(aggB,
                                              WtB + (size_t)(k + 1) * WSZ,
                                              convb + (size_t)k * NHID,
                                              gnn, nullptr,
                                              N_NODE, NHID, NHID, 1);
  }

  // 4) residue concat + reverse_features
  const int out_f4 = (5120 * 4608) / 4;       // 5,898,240
  write_out<<<out_f4 / 256, 256, 0, stream>>>(feats, x1, gnn, out);
}
